// CompatStatefulRes12SRWMModel_38740605010048
// MI455X (gfx1250) — compile-verified, matrix operand order checked by reading, not run on hardware
//
#include <hip/hip_runtime.h>
#include <math.h>

// ---------------------------------------------------------------------------
// Types for CDNA5 WMMA (wave32)
// ---------------------------------------------------------------------------
typedef __attribute__((ext_vector_type(16))) __bf16 v16bf;
typedef __attribute__((ext_vector_type(8)))  float  v8f;

__device__ __forceinline__ __bf16 bf0() {
  unsigned short z = 0;
  return __builtin_bit_cast(__bf16, z);
}
__device__ __forceinline__ __bf16 f2bf(float f) {
  unsigned u = __builtin_bit_cast(unsigned, f);
  unsigned r = u + 0x7FFFu + ((u >> 16) & 1u);   // round-to-nearest-even
  unsigned short h = (unsigned short)(r >> 16);
  return __builtin_bit_cast(__bf16, h);
}

// ---------------------------------------------------------------------------
// CDNA5 async copy global->LDS (ASYNCcnt-tracked DMA, bypasses VGPRs).
//   lds_off : LDS byte address (low 32 bits of generic pointer)
//   g       : global source (16B aligned), copies 16 bytes per call
// ---------------------------------------------------------------------------
__device__ __forceinline__ void async_b128(unsigned lds_off, const void* g) {
  asm volatile("global_load_async_to_lds_b128 %0, %1, off"
               :: "v"(lds_off), "v"((unsigned long long)(size_t)g)
               : "memory");
}
__device__ __forceinline__ void wait_async0() {
  asm volatile("s_wait_asynccnt 0x0" ::: "memory");
}
__device__ __forceinline__ unsigned lds_addr(const void* p) {
  return (unsigned)(size_t)p;   // LDS aperture: addr[31:0] is the LDS offset
}

// ---------------------------------------------------------------------------
// Small elementwise kernels
// ---------------------------------------------------------------------------
__global__ void k_convert_bf16(const float* __restrict__ in, __bf16* __restrict__ out, long n) {
  long i = (long)blockIdx.x * blockDim.x + threadIdx.x;
  long stride = (long)gridDim.x * blockDim.x;
  for (; i < n; i += stride) out[i] = f2bf(in[i]);
}

// Conv weights [Cout][ktot] fp32 -> bf16 [Cout][Kpad], K zero-padded to 32.
__global__ void k_pack_convw(const float* __restrict__ W, __bf16* __restrict__ out,
                             long Cout, long ktot, long Kpad) {
  long n = Cout * Kpad;
  long idx = (long)blockIdx.x * blockDim.x + threadIdx.x;
  long stride = (long)gridDim.x * blockDim.x;
  for (; idx < n; idx += stride) {
    long co = idx / Kpad, k = idx % Kpad;
    out[idx] = (k < ktot) ? f2bf(W[co * ktot + k]) : bf0();
  }
}

// W stored [O][I] fp32 -> out[i*O + o] bf16 (transposed to [K=I][N=O])
__global__ void k_pack_transpose(const float* __restrict__ W, __bf16* __restrict__ out,
                                 int O, int I) {
  long n = (long)O * I;
  long idx = (long)blockIdx.x * blockDim.x + threadIdx.x;
  long stride = (long)gridDim.x * blockDim.x;
  for (; idx < n; idx += stride) {
    long o = idx / I, i = idx % I;
    out[i * (long)O + o] = f2bf(W[idx]);
  }
}

__global__ void k_add_relu(float* __restrict__ A, const float* __restrict__ B, long n) {
  long i = (long)blockIdx.x * blockDim.x + threadIdx.x;
  long stride = (long)gridDim.x * blockDim.x;
  for (; i < n; i += stride) A[i] = fmaxf(A[i] + B[i], 0.f);
}

__global__ void k_add(float* __restrict__ A, const float* __restrict__ B, long n) {
  long i = (long)blockIdx.x * blockDim.x + threadIdx.x;
  long stride = (long)gridDim.x * blockDim.x;
  for (; i < n; i += stride) A[i] += B[i];
}

// ---------------------------------------------------------------------------
// Implicit-GEMM conv (3x3 pad1 or 1x1) via bf16 WMMA. One wave per 16x16 tile.
//   X  : bf16 NCHW input
//   Wp : bf16 packed weights [Cout][Kpad] (zero-padded K, 16B-aligned rows)
//   O  : fp32 NCHW output
//   GEMM view: M=Cout, N=Nimg*H*W, K=Cin*KHW*KHW (padded to Kpad)
//   A tile staged by async DMA to LDS; B tile is im2col gather.
// ---------------------------------------------------------------------------
template <int KHW>
__global__ void k_conv_wmma(const __bf16* __restrict__ X, const __bf16* __restrict__ Wp,
                            float* __restrict__ O, int Cin, int Cout, int H, int Wd,
                            int Nimg, int Kpad) {
  const int ktot = Cin * KHW * KHW;
  const long n0  = (long)blockIdx.x * 16;   // position tile
  const int co0  = blockIdx.y * 16;         // output-channel tile
  const int lane = threadIdx.x;             // 0..31 (wave32)
  const long HW  = (long)H * Wd;

  __shared__ alignas(16) __bf16 Asm[16 * 32];   // [m][k]
  __shared__ alignas(16) __bf16 Bsm[32 * 16];   // [k][n]

  v8f acc = {};

  // per-lane A-tile DMA geometry: row m = lane>>1, 16-elem segment = lane&1
  const int am  = lane >> 1;
  const int as  = (lane & 1) * 16;
  const unsigned aoff = lds_addr(&Asm[am * 32 + as]);

  // per-lane B-tile gather geometry: k-row = lane
  // decompose starting position once, then advance with carries
  long img0 = n0 / HW;
  long rem0 = n0 % HW;
  const int oy0 = (int)(rem0 / Wd);
  const int ox0 = (int)(rem0 % Wd);

  for (int k0 = 0; k0 < Kpad; k0 += 32) {
    // ---- A tile: async DMA, guard-free (K zero-padded) ----
    const __bf16* gA = Wp + (long)(co0 + am) * Kpad + k0 + as;
    async_b128(aoff, gA);
    async_b128(aoff + 16, gA + 8);
    if (k0 + 32 < Kpad)
      __builtin_prefetch(Wp + (long)(co0 + am) * Kpad + k0 + 32 + as, 0, 0);

    // ---- B tile: im2col gather (this lane owns k-row `lane`) ----
    {
      const int k = k0 + lane;
      int ci = 0, dy = 0, dx = 0;
      const bool kok = (k < ktot);
      if (kok) {
        ci = k / (KHW * KHW);
        int r = k % (KHW * KHW);
        dy = r / KHW - (KHW >> 1);
        dx = r % KHW - (KHW >> 1);
      }
      long img = img0;
      int oy = oy0, ox = ox0;
      const __bf16* Xc = X + ((img * Cin + ci) * (long)H) * Wd;
#pragma unroll 4
      for (int nn = 0; nn < 16; ++nn) {
        __bf16 v = bf0();
        if (kok) {
          int iy = oy + dy, ix = ox + dx;
          if (iy >= 0 && iy < H && ix >= 0 && ix < Wd)
            v = Xc[(long)iy * Wd + ix];
        }
        Bsm[lane * 16 + nn] = v;
        if (++ox == Wd) {                  // advance position with carries
          ox = 0;
          if (++oy == H) { oy = 0; ++img; Xc += (long)Cin * HW; }
        }
      }
    }
    wait_async0();   // A tile resident in LDS (single wave -> no barrier needed)

    // ---- build WMMA fragments per ISA VGPR layout ----
    v16bf a, b;
    const int mr = lane & 15;
    const int kb = (lane >> 4) * 8;
#pragma unroll
    for (int i = 0; i < 8; ++i) {
      int kk = kb + ((i < 4) ? 2 * i : 2 * i + 8);
      a[2 * i]     = Asm[mr * 32 + kk];
      a[2 * i + 1] = Asm[mr * 32 + kk + 1];
      b[2 * i]     = Bsm[kk * 16 + mr];
      b[2 * i + 1] = Bsm[(kk + 1) * 16 + mr];
    }
    acc = __builtin_amdgcn_wmma_f32_16x16x32_bf16(false, a, false, b, (short)0, acc,
                                                  false, false);
  }

  // ---- store D: lane holds (M = i + 8*(lane>>4), N = lane&15) ----
  const long n = n0 + (lane & 15);
  const long img = n / HW, rem = n % HW;
#pragma unroll
  for (int i = 0; i < 8; ++i) {
    int co = co0 + i + 8 * (lane >> 4);
    O[(img * Cout + co) * HW + rem] = acc[i];
  }
}

// ---------------------------------------------------------------------------
// bf16 WMMA GEMM: C[M][N] = A[M][K]*B[K][N] (+bias, +accum, relu)
// Requires M,N multiples of 16 and K multiple of 32 (true for all FFN calls).
// Both tiles staged by async DMA. grid: (N/16, M/16), one wave per block.
// ---------------------------------------------------------------------------
__global__ void k_gemm_wmma(const __bf16* __restrict__ A, const __bf16* __restrict__ Bm,
                            const float* __restrict__ bias, float* __restrict__ Cmat,
                            __bf16* __restrict__ bfout, int M, int N, int K,
                            int relu, int accum) {
  const int n0 = blockIdx.x * 16;
  const int m0 = blockIdx.y * 16;
  const int lane = threadIdx.x;

  __shared__ alignas(16) __bf16 Asm[16 * 32];
  __shared__ alignas(16) __bf16 Bsm[32 * 16];

  v8f acc = {};

  const int am = lane >> 1;
  const int as = (lane & 1) * 16;
  const unsigned aoff = lds_addr(&Asm[am * 32 + as]);
  const unsigned boff = lds_addr(&Bsm[lane * 16]);

  for (int k0 = 0; k0 < K; k0 += 32) {
    const __bf16* gA = A + (long)(m0 + am) * K + k0 + as;
    async_b128(aoff, gA);
    async_b128(aoff + 16, gA + 8);
    const __bf16* gB = Bm + (long)(k0 + lane) * N + n0;   // 16 bf16 per k-row
    async_b128(boff, gB);
    async_b128(boff + 16, gB + 8);
    wait_async0();

    v16bf a, b;
    const int mr = lane & 15;
    const int kb = (lane >> 4) * 8;
#pragma unroll
    for (int i = 0; i < 8; ++i) {
      int kk = kb + ((i < 4) ? 2 * i : 2 * i + 8);
      a[2 * i]     = Asm[mr * 32 + kk];
      a[2 * i + 1] = Asm[mr * 32 + kk + 1];
      b[2 * i]     = Bsm[kk * 16 + mr];
      b[2 * i + 1] = Bsm[(kk + 1) * 16 + mr];
    }
    acc = __builtin_amdgcn_wmma_f32_16x16x32_bf16(false, a, false, b, (short)0, acc,
                                                  false, false);
  }

  const int nn = lane & 15;
#pragma unroll
  for (int i = 0; i < 8; ++i) {
    int m = m0 + i + 8 * (lane >> 4);
    int n = n0 + nn;
    float v = acc[i] + bias[n];
    if (accum) v += Cmat[(long)m * N + n];
    if (relu) v = fmaxf(v, 0.f);
    Cmat[(long)m * N + n] = v;
    if (bfout) bfout[(long)m * N + n] = f2bf(v);
  }
}

// ---------------------------------------------------------------------------
// Train-mode BatchNorm: per-channel stats over (N,H,W), then normalize.
// ---------------------------------------------------------------------------
__global__ void k_bn_stats(const float* __restrict__ X, float* __restrict__ mean,
                           float* __restrict__ rstd, int C, int Nimg, long HW) {
  const int c = blockIdx.x;
  const int tid = threadIdx.x;
  __shared__ float s1[256], s2[256];
  float a = 0.f, b = 0.f;
  const long total = (long)Nimg * HW;
  for (long e = tid; e < total; e += 256) {
    long img = e / HW, p = e % HW;
    float v = X[(img * C + c) * HW + p];
    a += v; b += v * v;
  }
  s1[tid] = a; s2[tid] = b;
  __syncthreads();
  for (int s = 128; s > 0; s >>= 1) {
    if (tid < s) { s1[tid] += s1[tid + s]; s2[tid] += s2[tid + s]; }
    __syncthreads();
  }
  if (tid == 0) {
    float m = s1[0] / (float)total;
    float v = s2[0] / (float)total - m * m;
    mean[c] = m;
    rstd[c] = rsqrtf(v + 1e-5f);
  }
}

__global__ void k_bn_apply(float* __restrict__ X, const float* __restrict__ mean,
                           const float* __restrict__ rstd, const float* __restrict__ g,
                           const float* __restrict__ bb, int C, long HW, long total,
                           int relu, __bf16* __restrict__ bfout) {
  long i = (long)blockIdx.x * blockDim.x + threadIdx.x;
  long stride = (long)gridDim.x * blockDim.x;
  for (; i < total; i += stride) {
    int c = (int)((i / HW) % C);
    float v = (X[i] - mean[c]) * rstd[c] * g[c] + bb[c];
    if (relu) v = fmaxf(v, 0.f);
    X[i] = v;
    if (bfout) bfout[i] = f2bf(v);
  }
}

// ---------------------------------------------------------------------------
// 2x2 max-pool stride 2 (VALID); writes fp32 + bf16 copies.
// ---------------------------------------------------------------------------
__global__ void k_maxpool2(const float* __restrict__ X, float* __restrict__ Yf,
                           __bf16* __restrict__ Yb, int C, int H, int W, int Nimg) {
  const int H2 = H / 2, W2 = W / 2;
  const long total = (long)Nimg * C * H2 * W2;
  long i = (long)blockIdx.x * blockDim.x + threadIdx.x;
  long stride = (long)gridDim.x * blockDim.x;
  for (; i < total; i += stride) {
    int x2 = (int)(i % W2);
    int y2 = (int)((i / W2) % H2);
    int c  = (int)((i / ((long)W2 * H2)) % C);
    long img = i / ((long)W2 * H2 * C);
    const float* p = X + ((img * C + c) * (long)H + 2 * y2) * W + 2 * x2;
    float m = fmaxf(fmaxf(p[0], p[1]), fmaxf(p[W], p[W + 1]));
    Yf[i] = m;
    Yb[i] = f2bf(m);
  }
}

// Global average pool over 5x5 -> feat[128][256]
__global__ void k_avgpool(const float* __restrict__ X, float* __restrict__ feat) {
  int idx = blockIdx.x * blockDim.x + threadIdx.x;
  if (idx < 128 * 256) {
    const float* p = X + (long)idx * 25;
    float s = 0.f;
#pragma unroll
    for (int i = 0; i < 25; ++i) s += p[i];
    feat[idx] = s * (1.f / 25.f);
  }
}

// Projection: out[t][h] = feat[t]·W[h][:256] + W[h][256+fb[t]] + b[h]
__global__ void k_proj(const float* __restrict__ feat, const int* __restrict__ fb,
                       const float* __restrict__ W, const float* __restrict__ b,
                       float* __restrict__ out) {
  int t = blockIdx.x, h = threadIdx.x;   // 128 x 256
  const float* wr = W + (long)h * 261;
  float s = b[h];
  for (int c = 0; c < 256; ++c) s += wr[c] * feat[t * 256 + c];
  s += wr[256 + fb[t]];
  out[t * 256 + h] = s;
}

// LayerNorm over 256 features per token; optional bf16 copy.
__global__ void k_ln(const float* __restrict__ X, const float* __restrict__ g,
                     const float* __restrict__ bb, float* __restrict__ Y,
                     __bf16* __restrict__ Yb) {
  int t = blockIdx.x, h = threadIdx.x;   // 128 x 256
  __shared__ float red[256];
  float v = X[t * 256 + h];
  red[h] = v;
  __syncthreads();
  for (int s = 128; s > 0; s >>= 1) { if (h < s) red[h] += red[h + s]; __syncthreads(); }
  float mean = red[0] * (1.f / 256.f);
  __syncthreads();
  float d = v - mean;
  red[h] = d * d;
  __syncthreads();
  for (int s = 128; s > 0; s >>= 1) { if (h < s) red[h] += red[h + s]; __syncthreads(); }
  float var = red[0] * (1.f / 256.f);
  float o = g[h] * d * rsqrtf(var + 1e-5f) + bb[h];
  Y[t * 256 + h] = o;
  if (Yb) Yb[t * 256 + h] = f2bf(o);
}

// ---------------------------------------------------------------------------
// SRWM recurrence: one wave per (b,h); state matrices live in LDS for 16 steps.
// ---------------------------------------------------------------------------
__device__ __forceinline__ void softmax16(const float* in, float* out) {
  float m = in[0];
  for (int i = 1; i < 16; ++i) m = fmaxf(m, in[i]);
  float s = 0.f;
  for (int i = 0; i < 16; ++i) { float e = expf(in[i] - m); out[i] = e; s += e; }
  float inv = 1.f / s;
  for (int i = 0; i < 16; ++i) out[i] *= inv;
}

__global__ void k_srwm(const float* __restrict__ z, const float* __restrict__ pWy,
                       const float* __restrict__ pWq, const float* __restrict__ pWk,
                       const float* __restrict__ pwb, float* __restrict__ ys,
                       float* __restrict__ oWy, float* __restrict__ oWq,
                       float* __restrict__ oWk, float* __restrict__ owb) {
  const int b = blockIdx.x >> 4;
  const int h = blockIdx.x & 15;
  const int tid = threadIdx.x;

  __shared__ float Wy[256], Wq[256], Wk[256], wb[64];
  __shared__ float x[16], y[16], q[16], k[16], qs[16], ks[16], yo[16];
  __shared__ float dy[16], dq[16], dk[16], db[4], beta[4];

  for (int i = tid; i < 256; i += 32) {
    Wy[i] = pWy[h * 256 + i];
    Wq[i] = pWq[h * 256 + i];
    Wk[i] = pWk[h * 256 + i];
  }
  for (int i = tid; i < 64; i += 32) wb[i] = pwb[h * 64 + i];
  __syncthreads();

  for (int t = 0; t < 16; ++t) {
    if (tid < 16) x[tid] = z[((long)t * 8 + b) * 256 + h * 16 + tid];
    __syncthreads();

    if (tid < 16) {
      float sy = 0.f, sq = 0.f, sk = 0.f;
      for (int j = 0; j < 16; ++j) {
        float xv = x[j];
        sy += Wy[tid * 16 + j] * xv;
        sq += Wq[tid * 16 + j] * xv;
        sk += Wk[tid * 16 + j] * xv;
      }
      y[tid] = sy; q[tid] = sq; k[tid] = sk;
    } else if (tid < 20) {
      int c = tid - 16;
      float s = 0.f;
      for (int j = 0; j < 16; ++j) s += wb[j * 4 + c] * x[j];
      beta[c] = 1.f / (1.f + expf(-s));
    }
    __syncthreads();

    if (tid == 0)      softmax16(q, qs);
    else if (tid == 1) softmax16(k, ks);
    else if (tid == 2) softmax16(y, yo);
    __syncthreads();

    if (tid < 16) {
      float a = 0.f, bq = 0.f, ck = 0.f;
      for (int j = 0; j < 16; ++j) {
        float d = qs[j] - ks[j];
        a  += Wy[tid * 16 + j] * d;
        bq += Wq[tid * 16 + j] * d;
        ck += Wk[tid * 16 + j] * d;
      }
      dy[tid] = a; dq[tid] = bq; dk[tid] = ck;
    } else if (tid < 20) {
      int c = tid - 16;
      float s = 0.f;
      for (int j = 0; j < 16; ++j) s += wb[j * 4 + c] * (qs[j] - ks[j]);
      db[c] = s;
    }
    __syncthreads();

    for (int i = tid; i < 256; i += 32) {
      int r = i >> 4, c = i & 15;
      float kc = ks[c];
      Wy[i] += beta[0] * dy[r] * kc;
      Wq[i] += beta[1] * dq[r] * kc;
      Wk[i] += beta[2] * dk[r] * kc;
    }
    for (int i = tid; i < 64; i += 32) {
      int j = i >> 2, c = i & 3;
      wb[i] += beta[3] * ks[j] * db[c];
    }
    if (tid < 16) ys[((long)t * 8 + b) * 256 + h * 16 + tid] = yo[tid];
    __syncthreads();
  }

  const long base = (long)(b * 16 + h);
  for (int i = tid; i < 256; i += 32) {
    oWy[base * 256 + i] = Wy[i];
    oWq[base * 256 + i] = Wq[i];
    oWk[base * 256 + i] = Wk[i];
  }
  for (int i = tid; i < 64; i += 32) owb[base * 64 + i] = wb[i];
}

// logits[t][c] = out[t]·W[c] + b[c]
__global__ void k_logits(const float* __restrict__ out, const float* __restrict__ W,
                         const float* __restrict__ b, float* __restrict__ lg) {
  int tid = blockIdx.x * blockDim.x + threadIdx.x;
  if (tid < 640) {
    int t = tid / 5, c = tid % 5;
    float s = b[c];
    for (int h = 0; h < 256; ++h) s += W[c * 256 + h] * out[t * 256 + h];
    lg[tid] = s;
  }
}

// ---------------------------------------------------------------------------
// Host orchestration
// ---------------------------------------------------------------------------
static inline unsigned nb(long n) { return (unsigned)((n + 255) / 256); }
static inline int kpad32(int k) { return (k + 31) & ~31; }

extern "C" void kernel_launch(void* const* d_in, const int* in_sizes, int n_in,
                              void* d_out, int out_size, void* d_ws, size_t ws_size,
                              hipStream_t stream) {
  (void)in_sizes; (void)n_in; (void)out_size; (void)ws_size;

  const float* x  = (const float*)d_in[0];
  const int*   fb = (const int*)d_in[1];
  auto P = [&](int i) { return (const float*)d_in[i]; };

  // ---- workspace carve (all segments 16B-aligned) ----
  const size_t MAXE = 28901376ull;   // 128*32*84*84 (largest activation)
  char* w = (char*)d_ws;
  float*  S1f   = (float*)w;  w += MAXE * 4;
  float*  S2f   = (float*)w;  w += MAXE * 4;
  __bf16* S3b   = (__bf16*)w; w += MAXE * 2;
  __bf16* S4b   = (__bf16*)w; w += MAXE * 2;
  __bf16* Wb    = (__bf16*)w; w += 1048576ull * 2;
  float*  statm = (float*)w;  w += 256 * 4;
  float*  statr = (float*)w;  w += 256 * 4;
  float*  featb = (float*)w;  w += 32768 * 4;
  float*  outb  = (float*)w;  w += 32768 * 4;
  float*  zbuf  = (float*)w;  w += 32768 * 4;
  float*  zfbuf = (float*)w;  w += 32768 * 4;
  float*  ysbuf = (float*)w;  w += 32768 * 4;
  float*  hidf  = (float*)w;  w += 131072 * 4;
  __bf16* tokbf = (__bf16*)w; w += 32768 * 2;
  __bf16* hidbf = (__bf16*)w; w += 131072 * 2;

  // ---- input -> bf16 ----
  const long xcount = 128L * 3 * 84 * 84;
  k_convert_bf16<<<nb(xcount), 256, 0, stream>>>(x, S3b, xcount);

  const int CHs[4] = {32, 64, 128, 256};
  const int Hs[4]  = {84, 42, 21, 10};
  int Cin = 3;

  for (int blk = 0; blk < 4; ++blk) {
    const int Cout = CHs[blk];
    const int H = Hs[blk];
    const long HW = (long)H * H;
    const long NHW = 128L * HW;
    const long tot = NHW * Cout;
    const int pb = 2 + 12 * blk;
    const int K3 = Cin * 9, K3p = kpad32(K3);
    const int KCp = kpad32(Cout * 9);
    const int K1p = kpad32(Cin);
    dim3 gconv((unsigned)(NHW / 16), (unsigned)(Cout / 16));

    // conv1 + BN + ReLU
    k_pack_convw<<<nb((long)Cout * K3p), 256, 0, stream>>>(P(pb + 0), Wb, Cout, K3, K3p);
    k_conv_wmma<3><<<gconv, 32, 0, stream>>>(S3b, Wb, S1f, Cin, Cout, H, H, 128, K3p);
    k_bn_stats<<<Cout, 256, 0, stream>>>(S1f, statm, statr, Cout, 128, HW);
    k_bn_apply<<<nb(tot), 256, 0, stream>>>(S1f, statm, statr, P(pb + 1), P(pb + 2), Cout, HW, tot, 1, S4b);

    // conv2 + BN + ReLU
    k_pack_convw<<<nb((long)Cout * KCp), 256, 0, stream>>>(P(pb + 3), Wb, Cout, Cout * 9, KCp);
    k_conv_wmma<3><<<gconv, 32, 0, stream>>>(S4b, Wb, S2f, Cout, Cout, H, H, 128, KCp);
    k_bn_stats<<<Cout, 256, 0, stream>>>(S2f, statm, statr, Cout, 128, HW);
    k_bn_apply<<<nb(tot), 256, 0, stream>>>(S2f, statm, statr, P(pb + 4), P(pb + 5), Cout, HW, tot, 1, S4b);

    // conv3 + BN (no relu)
    k_pack_convw<<<nb((long)Cout * KCp), 256, 0, stream>>>(P(pb + 6), Wb, Cout, Cout * 9, KCp);
    k_conv_wmma<3><<<gconv, 32, 0, stream>>>(S4b, Wb, S1f, Cout, Cout, H, H, 128, KCp);
    k_bn_stats<<<Cout, 256, 0, stream>>>(S1f, statm, statr, Cout, 128, HW);
    k_bn_apply<<<nb(tot), 256, 0, stream>>>(S1f, statm, statr, P(pb + 7), P(pb + 8), Cout, HW, tot, 0, (__bf16*)nullptr);

    // shortcut 1x1 conv + BN (no relu)
    k_pack_convw<<<nb((long)Cout * K1p), 256, 0, stream>>>(P(pb + 9), Wb, Cout, Cin, K1p);
    k_conv_wmma<1><<<gconv, 32, 0, stream>>>(S3b, Wb, S2f, Cin, Cout, H, H, 128, K1p);
    k_bn_stats<<<Cout, 256, 0, stream>>>(S2f, statm, statr, Cout, 128, HW);
    k_bn_apply<<<nb(tot), 256, 0, stream>>>(S2f, statm, statr, P(pb + 10), P(pb + 11), Cout, HW, tot, 0, (__bf16*)nullptr);

    // residual add + relu, then 2x2 maxpool -> next block input
    k_add_relu<<<nb(tot), 256, 0, stream>>>(S1f, S2f, tot);
    const int H2 = H / 2;
    const long tot2 = 128L * Cout * H2 * H2;
    k_maxpool2<<<nb(tot2), 256, 0, stream>>>(S1f, S2f, S3b, Cout, H, H, 128);

    Cin = Cout;
  }

  // ---- head: global avg pool + projection ----
  k_avgpool<<<nb(32768), 256, 0, stream>>>(S2f, featb);
  k_proj<<<128, 256, 0, stream>>>(featb, fb, P(50), P(51), outb);

  // ---- SRWM transformer layers ----
  float* dout = (float*)d_out;
  for (int l = 0; l < 2; ++l) {
    const int lb = 52 + 12 * l;
    k_ln<<<128, 256, 0, stream>>>(outb, P(lb + 0), P(lb + 1), zbuf, (__bf16*)nullptr);

    float* oWy = dout + 640 + (long)l * 32768;
    float* oWq = dout + 640 + 65536 + (long)l * 32768;
    float* oWk = dout + 640 + 131072 + (long)l * 32768;
    float* owb = dout + 640 + 196608 + (long)l * 8192;
    k_srwm<<<128, 32, 0, stream>>>(zbuf, P(lb + 2), P(lb + 3), P(lb + 4), P(lb + 5),
                                   ysbuf, oWy, oWq, oWk, owb);
    k_add<<<nb(32768), 256, 0, stream>>>(outb, ysbuf, 32768);

    // FFN
    k_ln<<<128, 256, 0, stream>>>(outb, P(lb + 6), P(lb + 7), zfbuf, tokbf);
    k_pack_transpose<<<nb(1024L * 256), 256, 0, stream>>>(P(lb + 8), Wb, 1024, 256);
    k_gemm_wmma<<<dim3(64, 8), 32, 0, stream>>>(tokbf, Wb, P(lb + 9), hidf, hidbf,
                                                128, 1024, 256, 1, 0);
    k_pack_transpose<<<nb(256L * 1024), 256, 0, stream>>>(P(lb + 10), Wb, 256, 1024);
    k_gemm_wmma<<<dim3(16, 8), 32, 0, stream>>>(hidbf, Wb, P(lb + 11), outb,
                                                (__bf16*)nullptr, 128, 256, 1024, 0, 1);
  }

  // ---- logits ----
  k_logits<<<1, 640, 0, stream>>>(outb, P(76), P(77), dout);
}
